// MultiHeadAttention_22282290332409
// MI455X (gfx1250) — matrix operand order repaired and checked by reading, not compiled, not run on hardware
//
#include <hip/hip_runtime.h>
#include <hip/hip_bf16.h>
#include <math.h>

// ---------------------------------------------------------------------------
// MHA block on MI455X (gfx1250). Pure-bf16 WMMA data path:
//   f32 inputs --(one-time convert)--> bf16 --> WMMA GEMMs + flash attention
// S=2048, E=4096, H=8, DK=512, DH=64. Softmax scale = 1/DK (per reference).
// Fragment loads are grouped ahead of WMMA clauses so one s_wait_dscnt covers
// a dense run of v_wmma issues.
// ---------------------------------------------------------------------------

#define S_DIM 2048
#define E_DIM 4096
#define H_DIM 8
#define DK_DIM 512
#define DH_DIM 64

typedef __attribute__((ext_vector_type(16))) __bf16    v16bf;
typedef __attribute__((ext_vector_type(8)))  __bf16    v8bf;
typedef __attribute__((ext_vector_type(4)))  __bf16    v4bf;
typedef __attribute__((ext_vector_type(8)))  float     v8f;
typedef __attribute__((ext_vector_type(4)))  float     f32x4;
typedef __attribute__((ext_vector_type(4)))  unsigned  u32x4;

__device__ __forceinline__ __bf16 f2bf(float f) {
    union { float f; unsigned u; } x; x.f = f;
    unsigned r = x.u + 0x7FFFu + ((x.u >> 16) & 1u);   // round-to-nearest-even
    union { unsigned short s; __bf16 b; } y; y.s = (unsigned short)(r >> 16);
    return y.b;
}

// Two contiguous 16-byte chunks -> one 16-element bf16 fragment.
__device__ __forceinline__ v16bf ld16(const __bf16* p0, const __bf16* p1) {
    v8bf lo = *(const v8bf*)p0;
    v8bf hi = *(const v8bf*)p1;
    return __builtin_shufflevector(lo, hi, 0, 1, 2, 3, 4, 5, 6, 7,
                                            8, 9, 10, 11, 12, 13, 14, 15);
}

__device__ __forceinline__ float rmax16(float v) {
#pragma unroll
    for (int m = 1; m < 16; m <<= 1) v = fmaxf(v, __shfl_xor(v, m, 32));
    return v;
}
__device__ __forceinline__ float rsum16(float v) {
#pragma unroll
    for (int m = 1; m < 16; m <<= 1) v += __shfl_xor(v, m, 32);
    return v;
}

// Intra-wave LDS ordering fence: DS ops from one wave complete in order
// (ISA §15.5 counter rules), so cross-lane LDS transposes within a wave only
// need a compiler scheduling barrier, not a workgroup barrier.
__device__ __forceinline__ void wave_lds_fence() {
#if defined(__has_builtin)
#if __has_builtin(__builtin_amdgcn_wave_barrier)
    __asm__ volatile("" ::: "memory");
    __builtin_amdgcn_wave_barrier();
    __asm__ volatile("" ::: "memory");
#else
    __syncthreads();
#endif
#else
    __syncthreads();
#endif
}

#define WMMA_BF16(a, b, c) \
    __builtin_amdgcn_wmma_f32_16x16x32_bf16(false, (a), false, (b), (short)0, (c), false, false)

// ---------------------------------------------------------------------------
// One-time f32 -> bf16 conversion (vectorized 4-wide). n must be /4.
// ---------------------------------------------------------------------------
__global__ __launch_bounds__(256)
void cvt_f32_to_bf16(const float* __restrict__ x, __bf16* __restrict__ y, int n4) {
    const int i = blockIdx.x * 256 + threadIdx.x;
    if (i < n4) {
        f32x4 f = ((const f32x4*)x)[i];
        v4bf o;
        o[0] = f2bf(f[0]); o[1] = f2bf(f[1]); o[2] = f2bf(f[2]); o[3] = f2bf(f[3]);
        ((v4bf*)y)[i] = o;
    }
}

// ---------------------------------------------------------------------------
// GEMM: C[M,N] = A[M,K](bf16) @ B[K,N](bf16) + bias[N] (+ res[M,N] f32)
// Block = 256 threads (8 waves). Block tile 128(M) x 128(N), K chunk 32,
// double-buffered LDS (one barrier per chunk; next chunk's global loads
// overlap current chunk's WMMAs). Wave w: rows (w&3)*32, cols (w>>2)*64 ->
// 2x4 WMMA tiles. B staged TRANSPOSED in LDS ([n][k]) so all fragments are
// contiguous b128 loads. All 6 fragments load before the 8-WMMA clause.
// ---------------------------------------------------------------------------
template <bool HAS_RES, bool OUT_BF16>
__global__ __launch_bounds__(256)
void gemm_bf16_wmma(const __bf16* __restrict__ A, const __bf16* __restrict__ B,
                    const float* __restrict__ bias, const float* __restrict__ res,
                    void* __restrict__ Cout, int M, int K, int N) {
    __shared__ __bf16 As[2][128 * 32];    // [m][k]
    __shared__ __bf16 Bs[2][128 * 32];    // transposed: [n][k]

    const int tid  = threadIdx.x;
    const int lane = tid & 31;
    const int wid  = tid >> 5;
    const int half = lane >> 4;           // 0: lanes 0-15, 1: lanes 16-31
    const int l16  = lane & 15;

    const int m0 = blockIdx.y * 128;
    const int n0 = blockIdx.x * 128;
    const int wm = wid & 3;               // M quarter (32 rows)
    const int wn = wid >> 2;              // N half (64 cols)

    v8f acc[2][4];
#pragma unroll
    for (int i = 0; i < 2; ++i)
#pragma unroll
        for (int j = 0; j < 4; ++j) acc[i][j] = (v8f){};

    // Staging: pure bf16 copies, vectorized 16B.
    const int arow = tid >> 1, ac0 = (tid & 1) * 16;           // A: 16 elems/thread
    const int bk   = tid >> 3, bn0 = (tid & 7) * 16;           // B: 16 elems/thread
    auto stageA = [&](int k0, int buf) {
        const u32x4* src = (const u32x4*)(A + (size_t)(m0 + arow) * K + k0 + ac0);
        u32x4* dst = (u32x4*)&As[buf][arow * 32 + ac0];
        dst[0] = src[0]; dst[1] = src[1];
    };
    auto stageB = [&](int k0, int buf) {   // read [k][n] coalesced, scatter to [n][k]
        union { u32x4 u[2]; __bf16 b[16]; } t;
        const u32x4* src = (const u32x4*)(B + (size_t)(k0 + bk) * N + n0 + bn0);
        t.u[0] = src[0]; t.u[1] = src[1];
#pragma unroll
        for (int i = 0; i < 16; ++i) Bs[buf][(bn0 + i) * 32 + bk] = t.b[i];
    };

    stageA(0, 0); stageB(0, 0);
    const int nch = K / 32;
    for (int c = 0; c < nch; ++c) {
        __syncthreads();                              // buf[c&1] ready; prior reads done
        if (c + 1 < nch) { stageA((c + 1) * 32, (c + 1) & 1); stageB((c + 1) * 32, (c + 1) & 1); }
        if (c + 2 < nch) {                            // gfx1250 global_prefetch_b8
            __builtin_prefetch(A + (size_t)(m0 + arow) * K + (c + 2) * 32 + ac0, 0, 1);
            __builtin_prefetch(B + (size_t)((c + 2) * 32 + bk) * N + n0 + bn0, 0, 1);
        }
        const int buf = c & 1;

        // Group ALL fragment loads, then a dense 8-WMMA clause.
        v16bf afrag[2], bfrag[4];
#pragma unroll
        for (int i = 0; i < 2; ++i) {
            const __bf16* ap = &As[buf][(wm * 32 + i * 16 + l16) * 32];
            afrag[i] = ld16(ap + half * 8, ap + 16 + half * 8);
        }
#pragma unroll
        for (int j = 0; j < 4; ++j) {
            const __bf16* bp = &Bs[buf][(wn * 64 + j * 16 + l16) * 32];
            bfrag[j] = ld16(bp + half * 8, bp + 16 + half * 8);
        }
#pragma unroll
        for (int j = 0; j < 4; ++j)
#pragma unroll
            for (int i = 0; i < 2; ++i)
                acc[i][j] = WMMA_BF16(afrag[i], bfrag[j], acc[i][j]);
    }

    // Epilogue (C/D layout: lane -> N, vgpr v -> M (+8 on lane half 1))
#pragma unroll
    for (int j = 0; j < 4; ++j) {
        const int n = n0 + wn * 64 + j * 16 + l16;
        const float bn = bias[n];
#pragma unroll
        for (int i = 0; i < 2; ++i) {
#pragma unroll
            for (int v = 0; v < 8; ++v) {
                const int m = m0 + wm * 32 + i * 16 + v + half * 8;
                float r = acc[i][j][v] + bn;
                if (HAS_RES) r += res[(size_t)m * N + n];
                if (OUT_BF16) ((__bf16*)Cout)[(size_t)m * N + n] = f2bf(r);
                else          ((float*)Cout)[(size_t)m * N + n] = r;
            }
        }
    }
}

// ---------------------------------------------------------------------------
// Flash-style causal attention, bf16 in / bf16 out. Grid: (S/128, H).
// Block = 256 (8 waves); wave w owns 16 query rows. 32-key tiles through LDS.
// scale = 1/DK (per reference, NOT 1/sqrt).
// ---------------------------------------------------------------------------
__global__ __launch_bounds__(256)
void mha_flash_attn(const __bf16* __restrict__ Qp, const __bf16* __restrict__ Kp,
                    const __bf16* __restrict__ Vp, __bf16* __restrict__ Out) {
    __shared__ __bf16 Ks[32 * 64];        // [t][d]
    __shared__ __bf16 Vt[64 * 32];        // transposed: [d][t]
    __shared__ __bf16 Pw[8][16 * 32];     // per-wave P scratch [m][t]

    const int tid  = threadIdx.x;
    const int lane = tid & 31;
    const int wid  = tid >> 5;
    const int half = lane >> 4;
    const int l16  = lane & 15;

    const int h     = blockIdx.y;
    const int qblk  = blockIdx.x * 128;
    const int q0    = qblk + wid * 16;
    const int qlast = qblk + 127;
    const float INVDK = 1.0f / (float)DK_DIM;

    // Q fragments (A-frag layout) once, from bf16 global, contiguous 16B loads.
    v16bf qfrag[2];
    {
        const __bf16* qrow = Qp + (size_t)(q0 + l16) * DK_DIM + h * DH_DIM;
#pragma unroll
        for (int c = 0; c < 2; ++c)
            qfrag[c] = ld16(qrow + c * 32 + half * 8, qrow + c * 32 + 16 + half * 8);
    }

    float rowM[8], rowL[8];
    v8f oacc[4] = {{}, {}, {}, {}};
#pragma unroll
    for (int v = 0; v < 8; ++v) { rowM[v] = -INFINITY; rowL[v] = 0.0f; }

    const int trow = tid >> 3, d0 = (tid & 7) * 8;   // staging assignment
    for (int t0 = 0; t0 <= qlast; t0 += 32) {
        // Cooperative stage: K as-is (b128 copy), V transposed (b16 scatter).
        {
            const __bf16* kp = Kp + (size_t)(t0 + trow) * DK_DIM + h * DH_DIM + d0;
            const __bf16* vp = Vp + (size_t)(t0 + trow) * DK_DIM + h * DH_DIM + d0;
            *(u32x4*)&Ks[trow * 64 + d0] = *(const u32x4*)kp;
            union { u32x4 u; __bf16 b[8]; } t; t.u = *(const u32x4*)vp;
#pragma unroll
            for (int i = 0; i < 8; ++i) Vt[(d0 + i) * 32 + trow] = t.b[i];
        }
        if (t0 + 32 <= qlast) {
            __builtin_prefetch(Kp + (size_t)(t0 + 32 + trow) * DK_DIM + h * DH_DIM + d0, 0, 1);
            __builtin_prefetch(Vp + (size_t)(t0 + 32 + trow) * DK_DIM + h * DH_DIM + d0, 0, 1);
        }
        __syncthreads();

        // Scores: load all 4 K fragments, then a dense 4-WMMA clause.
        v16bf kfrag[2][2];
#pragma unroll
        for (int j = 0; j < 2; ++j)
#pragma unroll
            for (int c = 0; c < 2; ++c) {
                const __bf16* kp = &Ks[(j * 16 + l16) * 64 + c * 32];
                kfrag[j][c] = ld16(kp + half * 8, kp + 16 + half * 8);
            }
        v8f sacc[2];
#pragma unroll
        for (int j = 0; j < 2; ++j) {
            v8f s = {};
            s = WMMA_BF16(qfrag[0], kfrag[j][0], s);
            s = WMMA_BF16(qfrag[1], kfrag[j][1], s);
            sacc[j] = s;
        }

        // Online softmax (rows: vgpr index v; cols: 16-lane groups)
#pragma unroll
        for (int v = 0; v < 8; ++v) {
            const int qg = q0 + v + half * 8;
            float v0 = sacc[0][v] * INVDK;
            float v1 = sacc[1][v] * INVDK;
            if (t0 + l16      > qg) v0 = -INFINITY;
            if (t0 + 16 + l16 > qg) v1 = -INFINITY;

            const float tmax = rmax16(fmaxf(v0, v1));
            const float newM = fmaxf(rowM[v], tmax);
            const float scl  = __expf(rowM[v] - newM);
            rowM[v] = newM;
            const float p0 = __expf(v0 - newM);
            const float p1 = __expf(v1 - newM);
            rowL[v] = rowL[v] * scl + rsum16(p0 + p1);
#pragma unroll
            for (int n = 0; n < 4; ++n) oacc[n][v] *= scl;

            const int r = v + half * 8;
            Pw[wid][r * 32 + l16]      = f2bf(p0);
            Pw[wid][r * 32 + 16 + l16] = f2bf(p1);
        }
        wave_lds_fence();   // Pw is per-wave; same-wave DS ops complete in order

        // P(16x32) @ V(32x64): load P + all 4 V fragments, then 4 WMMAs.
        v16bf pfrag, vfrag[4];
        {
            const __bf16* pp = &Pw[wid][l16 * 32];
            pfrag = ld16(pp + half * 8, pp + 16 + half * 8);
        }
#pragma unroll
        for (int n = 0; n < 4; ++n) {
            const __bf16* vp = &Vt[(n * 16 + l16) * 32];
            vfrag[n] = ld16(vp + half * 8, vp + 16 + half * 8);
        }
#pragma unroll
        for (int n = 0; n < 4; ++n)
            oacc[n] = WMMA_BF16(pfrag, vfrag[n], oacc[n]);

        __syncthreads();    // protect Ks/Vt before next stage
    }

    // Normalize, convert, store bf16: Out[s, h*64 + d]
#pragma unroll
    for (int v = 0; v < 8; ++v) {
        const int qg = q0 + v + half * 8;
        const float inv = 1.0f / rowL[v];
#pragma unroll
        for (int n = 0; n < 4; ++n)
            Out[(size_t)qg * DK_DIM + h * DH_DIM + n * 16 + l16] = f2bf(oacc[n][v] * inv);
    }
}

// ---------------------------------------------------------------------------
// Row LayerNorm over E features, eps = 1e-4. Grid = S rows, block = 256.
// ---------------------------------------------------------------------------
__global__ __launch_bounds__(256)
void mha_layernorm(const float* __restrict__ X, const float* __restrict__ gamma,
                   const float* __restrict__ beta, float* __restrict__ Out) {
    const int row = blockIdx.x;
    const int tid = threadIdx.x;
    const int lane = tid & 31, wid = tid >> 5;
    const float* x = X + (size_t)row * E_DIM;

    float s = 0.0f, ss = 0.0f;
    for (int i = tid; i < E_DIM; i += 256) {
        const float v = x[i];
        s += v; ss += v * v;
    }
#pragma unroll
    for (int m = 1; m < 32; m <<= 1) {
        s  += __shfl_xor(s, m, 32);
        ss += __shfl_xor(ss, m, 32);
    }
    __shared__ float ls[8], lss[8];
    __shared__ float smean, srstd;
    if (lane == 0) { ls[wid] = s; lss[wid] = ss; }
    __syncthreads();
    if (tid == 0) {
        float S = 0.0f, SS = 0.0f;
#pragma unroll
        for (int w = 0; w < 8; ++w) { S += ls[w]; SS += lss[w]; }
        const float mu  = S / (float)E_DIM;
        const float var = SS / (float)E_DIM - mu * mu;
        smean = mu;
        srstd = rsqrtf(var + 1e-4f);
    }
    __syncthreads();
    const float mu = smean, rstd = srstd;
    for (int i = tid; i < E_DIM; i += 256)
        Out[(size_t)row * E_DIM + i] = gamma[i] * (x[i] - mu) * rstd + beta[i];
}

// ---------------------------------------------------------------------------
extern "C" void kernel_launch(void* const* d_in, const int* in_sizes, int n_in,
                              void* d_out, int out_size, void* d_ws, size_t ws_size,
                              hipStream_t stream) {
    const float* q     = (const float*)d_in[0];
    const float* k     = (const float*)d_in[1];
    const float* v     = (const float*)d_in[2];
    const float* Wq    = (const float*)d_in[3];
    const float* bq    = (const float*)d_in[4];
    const float* Wk    = (const float*)d_in[5];
    const float* bk    = (const float*)d_in[6];
    const float* Wv    = (const float*)d_in[7];
    const float* bv    = (const float*)d_in[8];
    const float* Wfc   = (const float*)d_in[9];
    const float* bfc   = (const float*)d_in[10];
    const float* gamma = (const float*)d_in[11];
    const float* beta  = (const float*)d_in[12];
    float* out = (float*)d_out;

    const size_t SE  = (size_t)S_DIM * E_DIM;    // 8,388,608
    const size_t EDK = (size_t)E_DIM * DK_DIM;   // 2,097,152
    const size_t SDK = (size_t)S_DIM * DK_DIM;   // 1,048,576

    // Workspace carve: bf16 copies of activations/weights, bf16 intermediates,
    // f32 residual buffer.
    char* p = (char*)d_ws;
    auto abf = [&](size_t n) { __bf16* r = (__bf16*)p; p += n * sizeof(__bf16); return r; };
    __bf16* qb   = abf(SE);
    __bf16* kb   = abf(SE);
    __bf16* vb   = abf(SE);
    __bf16* Wqb  = abf(EDK);
    __bf16* Wkb  = abf(EDK);
    __bf16* Wvb  = abf(EDK);
    __bf16* Wfcb = abf(EDK);
    __bf16* Qp   = abf(SDK);
    __bf16* Kp   = abf(SDK);
    __bf16* Vp   = abf(SDK);
    __bf16* Ao   = abf(SDK);
    float*  Res  = (float*)p;

    dim3 blk(256);

    // One-time f32 -> bf16 conversions
    auto cvt = [&](const float* src, __bf16* dst, size_t n) {
        const int n4 = (int)(n / 4);
        cvt_f32_to_bf16<<<dim3((n4 + 255) / 256), blk, 0, stream>>>(src, dst, n4);
    };
    cvt(q, qb, SE);  cvt(k, kb, SE);  cvt(v, vb, SE);
    cvt(Wq, Wqb, EDK); cvt(Wk, Wkb, EDK); cvt(Wv, Wvb, EDK); cvt(Wfc, Wfcb, EDK);

    // Projections: [2048,4096]@[4096,512] + bias -> bf16
    dim3 gproj(DK_DIM / 128, S_DIM / 128);
    gemm_bf16_wmma<false, true><<<gproj, blk, 0, stream>>>(qb, Wqb, bq, nullptr, Qp,
                                                           S_DIM, E_DIM, DK_DIM);
    gemm_bf16_wmma<false, true><<<gproj, blk, 0, stream>>>(kb, Wkb, bk, nullptr, Kp,
                                                           S_DIM, E_DIM, DK_DIM);
    gemm_bf16_wmma<false, true><<<gproj, blk, 0, stream>>>(vb, Wvb, bv, nullptr, Vp,
                                                           S_DIM, E_DIM, DK_DIM);

    // Causal attention (scale 1/DK), all heads -> bf16
    dim3 gattn(S_DIM / 128, H_DIM);
    mha_flash_attn<<<gattn, blk, 0, stream>>>(Qp, Kp, Vp, Ao);

    // FC: [2048,512]@[512,4096] + bias + residual(v, f32) -> f32
    dim3 gfc(E_DIM / 128, S_DIM / 128);
    gemm_bf16_wmma<true, false><<<gfc, blk, 0, stream>>>(Ao, Wfcb, bfc, v, Res,
                                                         S_DIM, DK_DIM, E_DIM);

    // LayerNorm rows
    mha_layernorm<<<dim3(S_DIM), blk, 0, stream>>>(Res, gamma, beta, out);
}
